// QANetAttBlock_15436112461930
// MI455X (gfx1250) — matrix operand order repaired and checked by reading, not compiled, not run on hardware
//
#include <hip/hip_runtime.h>

// Problem constants (match reference)
#define NB   16
#define NS   1024
#define NHID 512
#define NH   8
#define NDK  64
#define NEGV (-1.0e30f)

typedef __attribute__((ext_vector_type(16))) _Float16 v16h;
typedef __attribute__((ext_vector_type(8)))  float    v8f;

// ---------------------------------------------------------------------------
// CDNA5 async global->LDS copy (GLOBAL_LOAD_ASYNC_TO_LDS_B128, ASYNCcnt),
// guarded so the file still compiles on toolchains without the builtins.
// ---------------------------------------------------------------------------
#if __has_builtin(__builtin_amdgcn_global_load_async_to_lds_b128) && \
    __has_builtin(__builtin_amdgcn_s_wait_asynccnt)
#define HAVE_ASYNC_LDS 1
#endif

#define AS1 __attribute__((address_space(1)))
#define AS3 __attribute__((address_space(3)))
typedef int v4i_n __attribute__((vector_size(16)));  // native 16B vector (matches builtin)

__device__ __forceinline__ void async_copy16(const _Float16* gsrc, _Float16* ldst) {
#ifdef HAVE_ASYNC_LDS
  __builtin_amdgcn_global_load_async_to_lds_b128(
      (AS1 v4i_n*)const_cast<_Float16*>(gsrc), (AS3 v4i_n*)ldst, 0, 0);
#else
  *(int4*)ldst = *(const int4*)gsrc;
#endif
}

__device__ __forceinline__ void async_copy_wait() {
#ifdef HAVE_ASYNC_LDS
  __builtin_amdgcn_s_wait_asynccnt(0);
#endif
}

// ---------------------------------------------------------------------------
// WMMA helpers (gfx1250 wave32, V_WMMA_F32_16X16X32_F16)
// ---------------------------------------------------------------------------
__device__ __forceinline__ v8f wmma16(v16h a, v16h b, v8f c) {
  // 8 args: (neg_a, A, neg_b, B, c_mod, C, reuse_a, reuse_b)
  return __builtin_amdgcn_wmma_f32_16x16x32_f16(false, a, false, b, (short)0, c,
                                                false, false);
}

// A-fragment: 16x32 f16 tile, row-major with leading dim ld (halves).
// Lane layout: row m = lane&15; halves [0..15] = K in
// [half*16, half*16+16) (matches frag_BT).
// One contiguous run -> two 16B vector loads. Requires p 16B-aligned rows.
__device__ __forceinline__ v16h frag_A(const _Float16* p, int ld) {
  const int lane = threadIdx.x & 31;
  const int m = lane & 15, half = lane >> 4;
  const _Float16* s0 = p + m * ld + half * 16;
  v16h a;
  ((int4*)&a)[0] = ((const int4*)s0)[0];        // K = half*16 + 0..7
  ((int4*)&a)[1] = ((const int4*)(s0 + 8))[0];  // K = half*16 + 8..15
  return a;
}

// B-fragment: logical B[k][n] (32x16) read from a tile stored TRANSPOSED as
// [N][K] row-major with leading dim ld. Lane layout: col n = lane&15;
// halves [0..15] = K in [half*16, half*16+16) -> one contiguous 32B run.
__device__ __forceinline__ v16h frag_BT(const _Float16* p, int ld) {
  const int lane = threadIdx.x & 31;
  const int n = lane & 15, half = lane >> 4;
  const _Float16* s = p + n * ld + half * 16;
  v16h b;
  ((int4*)&b)[0] = ((const int4*)s)[0];
  ((int4*)&b)[1] = ((const int4*)s)[1];
  return b;
}

// ---------------------------------------------------------------------------
// Kernel 1: per-batch valid length + 1/sqrt(len)
// ---------------------------------------------------------------------------
__global__ void __launch_bounds__(256) len_kernel(const int* __restrict__ mask,
                                                  int* __restrict__ lens,
                                                  float* __restrict__ scales) {
  __shared__ int red[256];
  const int b = blockIdx.x, t = threadIdx.x;
  int s = 0;
  for (int i = t; i < NS; i += 256) s += mask[b * NS + i];
  red[t] = s;
  __syncthreads();
  for (int off = 128; off > 0; off >>= 1) {
    if (t < off) red[t] += red[t + off];
    __syncthreads();
  }
  if (t == 0) {
    lens[b] = red[0];
    scales[b] = rsqrtf((float)red[0]);
  }
}

// ---------------------------------------------------------------------------
// Kernel 2: fp32 -> f16 converts (with transposes so GEMM B operands are [N][K])
// ---------------------------------------------------------------------------
__global__ void __launch_bounds__(256) cvt_f32_f16(const float* __restrict__ src,
                                                   _Float16* __restrict__ dst, int n) {
  const int i = blockIdx.x * 256 + threadIdx.x;
  if (i < n) dst[i] = (_Float16)src[i];
}

// WcatT [3*HID][HID]: row j = (mat, head, d) output feature, col k = input feature.
__global__ void __launch_bounds__(256) build_wcatT(const float* __restrict__ Wq,
                                                   const float* __restrict__ Wk,
                                                   const float* __restrict__ Wv,
                                                   _Float16* __restrict__ WcatT) {
  const int i = blockIdx.x * 256 + threadIdx.x;
  if (i >= 3 * NHID * NHID) return;
  const int j = i / NHID;  // output feature 0..1535
  const int k = i % NHID;  // input feature
  const int mat = j >> 9;           // /512
  const int head = (j & 511) >> 6;  // /64
  const int d = j & 63;
  const float* W = (mat == 0) ? Wq : ((mat == 1) ? Wk : Wv);
  WcatT[i] = (_Float16)W[((size_t)head * NHID + k) * NDK + d];
}

// WoutT [HID][HID]: WoutT[j][k] = Wout[k][j]
__global__ void __launch_bounds__(256) build_woutT(const float* __restrict__ Wo,
                                                   _Float16* __restrict__ WoT) {
  const int i = blockIdx.x * 256 + threadIdx.x;
  if (i >= NHID * NHID) return;
  const int j = i / NHID, k = i % NHID;
  WoT[i] = (_Float16)Wo[(size_t)k * NHID + j];
}

// ---------------------------------------------------------------------------
// Shared WMMA GEMM main loop: C(128x64 block tile) = A[M x K] * B^T where the
// B operand is stored TRANSPOSED: Bt[N][K] row-major (ldb = K stride).
// f16 in, f32 accumulate. 8 waves: wave = (wn 0..1)*4 + (wm 0..3); each 32x32.
// K staged 64 wide -> 8 WMMAs per barrier pair.
// ---------------------------------------------------------------------------
#define LDA_S 72  // 64 + 8 pad (144B rows: 16B-aligned, conflict-breaking)
#define LDB_S 72

__device__ __forceinline__ void gemm_tile_loop(const _Float16* __restrict__ A, int lda,
                                               const _Float16* __restrict__ Bt, int ldb,
                                               int mBase, int nBase, int Kdim,
                                               _Float16* As, _Float16* Bs,
                                               v8f (&c)[2][2]) {
  const int wave = threadIdx.x >> 5;
  const int wm = (wave & 3) * 32;
  const int wn = (wave >> 2) * 32;
  for (int kBase = 0; kBase < Kdim; kBase += 64) {
    // Stage A tile 128x64 (four b128s/thread) and Bt tile 64(n)x64(k) (two b128s)
#pragma unroll
    for (int pass = 0; pass < 4; ++pass) {
      const int idx = (threadIdx.x + pass * 256) * 8;
      const int row = idx >> 6, col = idx & 63;
      async_copy16(&A[(size_t)(mBase + row) * lda + kBase + col],
                   &As[row * LDA_S + col]);
    }
#pragma unroll
    for (int pass = 0; pass < 2; ++pass) {
      const int idx = (threadIdx.x + pass * 256) * 8;
      const int row = idx >> 6, col = idx & 63;  // row = n, col = k
      async_copy16(&Bt[(size_t)(nBase + row) * ldb + kBase + col],
                   &Bs[row * LDB_S + col]);
    }
    async_copy_wait();
    __syncthreads();
#pragma unroll
    for (int kk = 0; kk < 64; kk += 32) {
      const v16h a0 = frag_A(As + (wm + 0) * LDA_S + kk, LDA_S);
      const v16h a1 = frag_A(As + (wm + 16) * LDA_S + kk, LDA_S);
      const v16h b0 = frag_BT(Bs + (wn + 0) * LDB_S + kk, LDB_S);
      const v16h b1 = frag_BT(Bs + (wn + 16) * LDB_S + kk, LDB_S);
      c[0][0] = wmma16(a0, b0, c[0][0]);
      c[0][1] = wmma16(a0, b1, c[0][1]);
      c[1][0] = wmma16(a1, b0, c[1][0]);
      c[1][1] = wmma16(a1, b1, c[1][1]);
    }
    __syncthreads();
  }
}

// ---------------------------------------------------------------------------
// Kernel 3: QKV projection. x_f16[b] (1024x512) @ WcatT^T -> Q/K f16 [B,H,S,DK],
// V f16 stored TRANSPOSED [B,H,DK,S] (feeds contiguous PV B-fragments).
// ---------------------------------------------------------------------------
__global__ void __launch_bounds__(256) proj_kernel(const _Float16* __restrict__ xh,
                                                   const _Float16* __restrict__ WcatT,
                                                   _Float16* __restrict__ Qb,
                                                   _Float16* __restrict__ Kb,
                                                   _Float16* __restrict__ Vb) {
  __shared__ _Float16 As[128 * LDA_S];
  __shared__ _Float16 Bs[64 * LDB_S];
  const int mBase = blockIdx.x * 128;
  const int nBase = blockIdx.y * 64;
  const int b = blockIdx.z;
  v8f c[2][2] = {};
  gemm_tile_loop(xh + (size_t)b * NS * NHID, NHID, WcatT, NHID, mBase, nBase, NHID,
                 As, Bs, c);

  const int mat = nBase / NHID;
  const int head = (nBase % NHID) / NDK;
  const int lane = threadIdx.x & 31;
  const int n = lane & 15, half = lane >> 4;
  const int wave = threadIdx.x >> 5;
  const int wm = (wave & 3) * 32, wn = (wave >> 2) * 32;
  if (mat == 2) {
    // V transposed: Vb[b,h,d,s]
    _Float16* dv = Vb + (size_t)(b * NH + head) * NDK * NS;
#pragma unroll
    for (int i = 0; i < 2; ++i)
#pragma unroll
      for (int j = 0; j < 2; ++j) {
        const int m0 = mBase + wm + i * 16 + half * 8;
        const int d = wn + j * 16 + n;
#pragma unroll
        for (int r = 0; r < 8; ++r)
          dv[(size_t)d * NS + (m0 + r)] = (_Float16)c[i][j][r];
      }
  } else {
    _Float16* dst = ((mat == 0) ? Qb : Kb) + (size_t)(b * NH + head) * NS * NDK;
#pragma unroll
    for (int i = 0; i < 2; ++i)
#pragma unroll
      for (int j = 0; j < 2; ++j) {
        const int m0 = mBase + wm + i * 16 + half * 8;
        const int d = wn + j * 16 + n;
#pragma unroll
        for (int r = 0; r < 8; ++r)
          dst[(size_t)(m0 + r) * NDK + d] = (_Float16)c[i][j][r];
      }
  }
}

// ---------------------------------------------------------------------------
// Kernel 4: flash-style attention per (qTile=128, head, batch).
// Wave owns 16 queries; streams 32-key K/V chunks through LDS.
// K tile stored [key][dk] (frag_BT contiguous); V tile stored [d][key]
// (transposed in global, frag_BT contiguous).
// ---------------------------------------------------------------------------
#define LDK  72  // 64 + 8 (144B rows)
#define LDVT 40  // 32 + 8 (80B rows)
#define LDP  40  // 32 + 8

__global__ void __launch_bounds__(256) attn_kernel(const _Float16* __restrict__ Qb,
                                                   const _Float16* __restrict__ Kb,
                                                   const _Float16* __restrict__ Vb,
                                                   const int* __restrict__ lens,
                                                   const float* __restrict__ scales,
                                                   _Float16* __restrict__ hbuf) {
  __shared__ _Float16 Kt[32 * LDK];    // [key][dk]
  __shared__ _Float16 Vt[64 * LDVT];   // [d][key]
  __shared__ _Float16 Pt[8][16 * LDP]; // wave-private P scratch

  const int head = blockIdx.y, b = blockIdx.z;
  const int len = lens[b];
  const float scale = scales[b];
  const size_t bh = (size_t)(b * NH + head) * NS * NDK;
  const _Float16* Q = Qb + bh;
  const _Float16* Kg = Kb + bh;
  const _Float16* Vg = Vb + bh;  // transposed [DK][NS]

  const int wave = threadIdx.x >> 5;
  const int lane = threadIdx.x & 31;
  const int n = lane & 15, half = lane >> 4;
  const int qBase = blockIdx.x * 128 + wave * 16;

  // Q fragments for this wave's 16 queries (dk 0..31 and 32..63), from global
  const v16h qa0 = frag_A(Q + (size_t)qBase * NDK, NDK);
  const v16h qa1 = frag_A(Q + (size_t)qBase * NDK + 32, NDK);

  float mrow[8], lrow[8];
  bool qvalid[8];
#pragma unroll
  for (int r = 0; r < 8; ++r) {
    mrow[r] = -3.0e38f;
    lrow[r] = 0.0f;
    qvalid[r] = (qBase + r + half * 8) < len;  // prefix mask
  }
  v8f o[4] = {};

  for (int kc = 0; kc < NS / 32; ++kc) {
    const int keyBase = kc * 32;
    {  // K chunk [32 key][64 dk]: one b128/thread
      const int idx = threadIdx.x * 8;
      const int row = idx >> 6, col = idx & 63;
      async_copy16(&Kg[(size_t)(keyBase + row) * NDK + col], &Kt[row * LDK + col]);
    }
    {  // V^T chunk [64 d][32 key]: one b128/thread
      const int idx = threadIdx.x * 8;
      const int row = idx >> 5, col = idx & 31;
      async_copy16(&Vg[(size_t)row * NS + keyBase + col], &Vt[row * LDVT + col]);
    }
    async_copy_wait();
    __syncthreads();

    // scores = Q(16x64) @ K_chunk^T(64x32): two 16-key subtiles x two dk steps
    v8f s0 = {}, s1 = {};
    s0 = wmma16(qa0, frag_BT(Kt, LDK), s0);
    s0 = wmma16(qa1, frag_BT(Kt + 32, LDK), s0);
    s1 = wmma16(qa0, frag_BT(Kt + 16 * LDK, LDK), s1);
    s1 = wmma16(qa1, frag_BT(Kt + 16 * LDK + 32, LDK), s1);

    // masked-scale + online softmax update (row = r + half*8, col = key at lane n)
    const int key0 = keyBase + n;
    const int key1 = keyBase + 16 + n;
    float p0[8], p1[8];
#pragma unroll
    for (int r = 0; r < 8; ++r) {
      float v0 = (qvalid[r] && key0 < len) ? s0[r] * scale : NEGV;
      float v1 = (qvalid[r] && key1 < len) ? s1[r] * scale : NEGV;
      float mx = fmaxf(v0, v1);
#pragma unroll
      for (int off = 1; off < 16; off <<= 1)
        mx = fmaxf(mx, __shfl_xor(mx, off, 32));  // within 16-lane half
      const float nm = fmaxf(mrow[r], mx);
      const float corr = __expf(mrow[r] - nm);
      const float e0 = __expf(v0 - nm);
      const float e1 = __expf(v1 - nm);
      float sum = e0 + e1;
#pragma unroll
      for (int off = 1; off < 16; off <<= 1) sum += __shfl_xor(sum, off, 32);
      lrow[r] = lrow[r] * corr + sum;
      mrow[r] = nm;
      p0[r] = e0;
      p1[r] = e1;
#pragma unroll
      for (int t = 0; t < 4; ++t) o[t][r] *= corr;
    }

    // C-layout -> A-fragment layout via wave-private LDS round trip
    _Float16* Pw = &Pt[wave][0];
#pragma unroll
    for (int r = 0; r < 8; ++r) {
      Pw[(r + half * 8) * LDP + n] = (_Float16)p0[r];
      Pw[(r + half * 8) * LDP + 16 + n] = (_Float16)p1[r];
    }
    const v16h pa = frag_A(Pw, LDP);  // 16 queries x 32 keys

    // O += P @ V_chunk: B[k=key][n=d] = Vt[d][key] -> frag_BT, contiguous
#pragma unroll
    for (int t = 0; t < 4; ++t)
      o[t] = wmma16(pa, frag_BT(Vt + (t * 16) * LDVT, LDVT), o[t]);

    __syncthreads();
  }

  // normalize and write h[b, q, head*64 + d] (f16)
#pragma unroll
  for (int r = 0; r < 8; ++r) {
    const float inv = 1.0f / lrow[r];
    const int q = qBase + r + half * 8;
#pragma unroll
    for (int t = 0; t < 4; ++t)
      hbuf[((size_t)b * NS + q) * NHID + head * NDK + t * 16 + n] =
          (_Float16)(o[t][r] * inv);
  }
}

// ---------------------------------------------------------------------------
// Kernel 5: output projection h(1024x512) @ Wout(512x512) -> fp32 out
// (Wout supplied transposed as WoT[N][K])
// ---------------------------------------------------------------------------
__global__ void __launch_bounds__(256) out_kernel(const _Float16* __restrict__ hbuf,
                                                  const _Float16* __restrict__ WoT,
                                                  float* __restrict__ out) {
  __shared__ _Float16 As[128 * LDA_S];
  __shared__ _Float16 Bs[64 * LDB_S];
  const int mBase = blockIdx.x * 128;
  const int nBase = blockIdx.y * 64;
  const int b = blockIdx.z;
  v8f c[2][2] = {};
  gemm_tile_loop(hbuf + (size_t)b * NS * NHID, NHID, WoT, NHID, mBase, nBase, NHID,
                 As, Bs, c);

  const int lane = threadIdx.x & 31;
  const int n = lane & 15, half = lane >> 4;
  const int wave = threadIdx.x >> 5;
  const int wm = (wave & 3) * 32, wn = (wave >> 2) * 32;
#pragma unroll
  for (int i = 0; i < 2; ++i)
#pragma unroll
    for (int j = 0; j < 2; ++j) {
      const int m0 = mBase + wm + i * 16 + half * 8;
      const int col = nBase + wn + j * 16 + n;
#pragma unroll
      for (int r = 0; r < 8; ++r)
        out[((size_t)b * NS + m0 + r) * NHID + col] = c[i][j][r];
    }
}

// ---------------------------------------------------------------------------
// Host launch
// ---------------------------------------------------------------------------
extern "C" void kernel_launch(void* const* d_in, const int* in_sizes, int n_in,
                              void* d_out, int out_size, void* d_ws, size_t ws_size,
                              hipStream_t stream) {
  (void)in_sizes; (void)n_in; (void)out_size; (void)ws_size;
  const float* x = (const float*)d_in[0];
  const int* mask = (const int*)d_in[1];
  const float* Wq = (const float*)d_in[2];
  const float* Wk = (const float*)d_in[3];
  const float* Wv = (const float*)d_in[4];
  const float* Wout = (const float*)d_in[5];
  float* out = (float*)d_out;

  // Workspace layout (256B aligned). hbuf aliases xh: xh is dead after proj_kernel.
  char* ws = (char*)d_ws;
  size_t off = 0;
  auto take = [&](size_t bytes) {
    size_t o = off;
    off = (off + bytes + 255) & ~(size_t)255;
    return o;
  };
  const size_t o_xh   = take((size_t)NB * NS * NHID * 2);   // also reused as hbuf
  const size_t o_wcat = take((size_t)3 * NHID * NHID * 2);
  const size_t o_wo   = take((size_t)NHID * NHID * 2);
  const size_t o_q    = take((size_t)NB * NH * NS * NDK * 2);
  const size_t o_k    = take((size_t)NB * NH * NS * NDK * 2);
  const size_t o_v    = take((size_t)NB * NH * NS * NDK * 2);
  const size_t o_len  = take(NB * sizeof(int));
  const size_t o_scl  = take(NB * sizeof(float));

  _Float16* xh    = (_Float16*)(ws + o_xh);
  _Float16* wcatT = (_Float16*)(ws + o_wcat);
  _Float16* woT   = (_Float16*)(ws + o_wo);
  _Float16* Qb    = (_Float16*)(ws + o_q);
  _Float16* Kb    = (_Float16*)(ws + o_k);
  _Float16* Vb    = (_Float16*)(ws + o_v);   // stored [B,H,DK,S]
  int* lens       = (int*)(ws + o_len);
  float* scales   = (float*)(ws + o_scl);
  _Float16* hbuf  = xh;  // alias: safe, xh consumed before attention writes h

  len_kernel<<<NB, 256, 0, stream>>>(mask, lens, scales);

  const int nx = NB * NS * NHID;
  cvt_f32_f16<<<(nx + 255) / 256, 256, 0, stream>>>(x, xh, nx);
  build_wcatT<<<(3 * NHID * NHID + 255) / 256, 256, 0, stream>>>(Wq, Wk, Wv, wcatT);
  build_woutT<<<(NHID * NHID + 255) / 256, 256, 0, stream>>>(Wout, woT);

  proj_kernel<<<dim3(NS / 128, (3 * NHID) / 64, NB), 256, 0, stream>>>(xh, wcatT, Qb,
                                                                       Kb, Vb);
  attn_kernel<<<dim3(NS / 128, NH, NB), 256, 0, stream>>>(Qb, Kb, Vb, lens, scales,
                                                          hbuf);
  out_kernel<<<dim3(NS / 128, NHID / 64, NB), 256, 0, stream>>>(hbuf, woT, out);
}
